// Attention_22849226014998
// MI455X (gfx1250) — matrix operand order repaired and checked by reading, not compiled, not run on hardware
//
#include <hip/hip_runtime.h>

// Problem constants (reference: B=2, H=16, L=2048, D=128)
#define BATCH 2
#define NH    16
#define SEQ   2048
#define HD    128
#define BQ    16            // q-rows per workgroup
#define NWAVE 8             // 256 threads, wave32
#define WCOLS (SEQ / NWAVE) // 256 k-columns per wave
#define NCB   (WCOLS / 16)  // 16 col-blocks of 16 per wave
#define SPAD  4             // LDS row pad (floats) to break bank conflicts

typedef __attribute__((ext_vector_type(16))) __bf16 v16bf;
typedef __attribute__((ext_vector_type(8)))  __bf16 v8bf;
typedef __attribute__((ext_vector_type(4)))  __bf16 v4bf;
typedef __attribute__((ext_vector_type(8)))  float  v8f;

// ---------------- pre-pass 1: fp32 -> bf16 bulk convert --------------------
__global__ __launch_bounds__(256)
void cvt_f32_to_bf16(const float4* __restrict__ src, v4bf* __restrict__ dst, int n4) {
  int i = blockIdx.x * 256 + threadIdx.x;
  if (i >= n4) return;
  float4 x = src[i];
  v4bf o;
  o[0] = (__bf16)x.x; o[1] = (__bf16)x.y; o[2] = (__bf16)x.z; o[3] = (__bf16)x.w;
  dst[i] = o;
}

// ---------------- pre-pass 2: K (bh,l,d) f32 -> K^T (bh,d,l) bf16 ----------
__global__ __launch_bounds__(256)
void cvt_transpose_bf16(const float* __restrict__ src, __bf16* __restrict__ dst) {
  __shared__ float tile[32][33];
  const int bh = blockIdx.z;
  const int l0 = blockIdx.x * 32;
  const int d0 = blockIdx.y * 32;
  const int tx = threadIdx.x & 31;
  const int ty = threadIdx.x >> 5;       // 0..7
  const float* s = src + ((size_t)bh * SEQ + l0) * HD + d0;
  #pragma unroll
  for (int i = 0; i < 4; ++i)
    tile[ty + i * 8][tx] = s[(size_t)(ty + i * 8) * HD + tx];
  __syncthreads();
  __bf16* dp = dst + ((size_t)bh * HD + d0) * SEQ + l0;
  #pragma unroll
  for (int i = 0; i < 4; ++i)
    dp[(size_t)(ty + i * 8) * SEQ + tx] = (__bf16)tile[tx][ty + i * 8];
}

// ---------------- pre-pass 3: fused additive bias (amortized 16x over H) ---
// bias[b,l,l'] = mask ? probs*alpha : -inf    (removes cmp/cndmask + one
// global stream from the O(L^2) hot loop of every head)
__global__ __launch_bounds__(256)
void build_bias(const float4* __restrict__ probs, const int4* __restrict__ mask,
                const float* __restrict__ alphaPtr, float4* __restrict__ bias, int n4) {
  int i = blockIdx.x * 256 + threadIdx.x;
  if (i >= n4) return;
  const float a = alphaPtr[0];
  const float ninf = -__builtin_inff();
  float4 p = probs[i];
  int4   m = mask[i];
  float4 o;
  o.x = m.x ? p.x * a : ninf;
  o.y = m.y ? p.y * a : ninf;
  o.z = m.z ? p.z * a : ninf;
  o.w = m.w ? p.w * a : ninf;
  bias[i] = o;
}

// ---------------- fused attention ------------------------------------------
// MODE 2: bf16 Q / K^T / V in scratch + fused bias array   (fast path)
// MODE 1: bf16 Q / K^T / V in scratch, raw probs+mask
// MODE 0: convert inline from fp32 (no scratch fallback)
template<int MODE>
__global__ __launch_bounds__(256)
void attn_fused(const float* __restrict__ qf, const float* __restrict__ kf,
                const float* __restrict__ vf,
                const __bf16* __restrict__ qbs, const __bf16* __restrict__ ktbs,
                const __bf16* __restrict__ vbs,
                const float* __restrict__ biasArr,
                const int* __restrict__ mask, const float* __restrict__ probs,
                const float* __restrict__ alphaPtr,
                float* __restrict__ outAttn, float* __restrict__ outP)
{
  __shared__ __bf16 sQ[BQ][HD + 8];        // ~4.3 KB bf16 Q tile
  __shared__ float  sS[BQ][SEQ + SPAD];    // ~128 KB scores -> exp(scores-max)
  __shared__ float  sRed[NWAVE][BQ];
  __shared__ float  sRowMax[BQ];
  __shared__ float  sRowSum[BQ];
  __shared__ float  sOut[BQ][HD];          // 8 KB cross-wave P@V accumulator

  const int t    = threadIdx.x;
  const int lane = t & 31;
  const int w    = t >> 5;
  const int qblk = blockIdx.x & (SEQ / BQ - 1);
  const int bh   = blockIdx.x >> 7;
  const int b    = bh >> 4;
  const int q0   = qblk * BQ;

  const float alpha = alphaPtr[0];
  const float scale = 0.08838834764831845f;  // 1/sqrt(128)

  const size_t headOff = (size_t)bh * SEQ * HD;

  // zero the output accumulator
  #pragma unroll
  for (int i = 0; i < (BQ * HD) / 256; ++i)
    ((float*)sOut)[t + i * 256] = 0.f;

  // ---- stage Q tile in LDS (bf16) ----
  if constexpr (MODE >= 1) {
    const __bf16* qp = qbs + headOff + (size_t)q0 * HD;
    int e = t * 8;
    int row = e >> 7, col = e & (HD - 1);
    *(v8bf*)&sQ[row][col] = *(const v8bf*)(qp + e);
  } else {
    const float4* q4 = (const float4*)(qf + headOff + (size_t)q0 * HD);
    int i4 = t * 2;
    float4 a = q4[i4], c2 = q4[i4 + 1];
    int e = i4 * 4;
    int row = e >> 7, col = e & (HD - 1);
    v8bf o;
    o[0] = (__bf16)a.x;  o[1] = (__bf16)a.y;  o[2] = (__bf16)a.z;  o[3] = (__bf16)a.w;
    o[4] = (__bf16)c2.x; o[5] = (__bf16)c2.y; o[6] = (__bf16)c2.z; o[7] = (__bf16)c2.w;
    *(v8bf*)&sQ[row][col] = o;
  }
  __syncthreads();

  // ---- Q A-fragments (ISA 16-bit A layout: lane=M row, split-K in lane) ----
  const int mrow = lane & 15;
  const int kbo  = (lane >> 4) << 3;   // A-layout K sub-offset 0 or 8
  const int rhi  = (lane >> 4) << 3;   // C-layout row offset 0 or 8
  v16bf aq[4];
  #pragma unroll
  for (int c = 0; c < 4; ++c) {
    v8bf lo = *(const v8bf*)&sQ[mrow][c * 32 + kbo];
    v8bf hi = *(const v8bf*)&sQ[mrow][c * 32 + 16 + kbo];
    aq[c] = __builtin_shufflevector(lo, hi, 0,1,2,3,4,5,6,7,8,9,10,11,12,13,14,15);
  }

  // ---- S = (Q K^T)*scale + bias ----
  // B layout (ISA): lane = N column of B, 16 K values within lane (table order).
  float rmax[8];
  #pragma unroll
  for (int g = 0; g < 8; ++g) rmax[g] = -__builtin_inff();

  for (int cb = 0; cb < NCB; ++cb) {
    const int n0   = w * WCOLS + cb * 16;   // k-column base of this tile
    const int kcol = n0 + mrow;             // C-layout column for epilogue

    v8f acc = {};
    #pragma unroll
    for (int c = 0; c < 4; ++c) {
      v16bf bk;
      if constexpr (MODE >= 1) {
        // K^T (bh, d, l): lane's B-column kcol = n0+mrow; d in table order (kbo split)
        const __bf16* kt = ktbs + headOff + (size_t)(c * 32 + kbo) * SEQ + n0 + mrow;
        if (c == 0 && cb + 1 < NCB) __builtin_prefetch(kt + 16, 0, 1);
        #pragma unroll
        for (int j = 0; j < 16; ++j)
          bk[j] = kt[(size_t)(j < 8 ? j : j + 8) * SEQ];
      } else {
        const float* kcolp = kf + headOff + (size_t)(n0 + mrow) * HD + (c * 32 + kbo);
        #pragma unroll
        for (int j = 0; j < 16; ++j)
          bk[j] = (__bf16)kcolp[j < 8 ? j : j + 8];   // gather fallback
      }
      acc = __builtin_amdgcn_wmma_f32_16x16x32_bf16(false, aq[c], false, bk,
                                                    (short)0, acc, false, false);
    }
    // fused epilogue: scale + bias, row-max tracking, stash to LDS
    #pragma unroll
    for (int g = 0; g < 8; ++g) {
      int row  = g + rhi;
      int qrow = q0 + row;
      size_t pidx = ((size_t)b * SEQ + qrow) * SEQ + kcol;
      float s;
      if constexpr (MODE == 2) {
        s = acc[g] * scale + biasArr[pidx];          // single load, single FMA
      } else {
        s = acc[g] * scale + probs[pidx] * alpha;
        if (mask[pidx] == 0) s = -__builtin_inff();
      }
      rmax[g] = fmaxf(rmax[g], s);
      sS[row][kcol] = s;
    }
  }

  // ---- global row max: shuffle within 16-lane groups, then across waves ----
  #pragma unroll
  for (int g = 0; g < 8; ++g) {
    float m = rmax[g];
    #pragma unroll
    for (int off = 8; off >= 1; off >>= 1)
      m = fmaxf(m, __shfl_xor(m, off, 32));
    rmax[g] = m;
  }
  if (mrow == 0) {
    #pragma unroll
    for (int g = 0; g < 8; ++g) sRed[w][g + rhi] = rmax[g];
  }
  __syncthreads();
  if (t < BQ) {
    float m = sRed[0][t];
    #pragma unroll
    for (int ww = 1; ww < NWAVE; ++ww) m = fmaxf(m, sRed[ww][t]);
    sRowMax[t] = m;
  }
  __syncthreads();

  // ---- exp pass (unnormalized) + row sums ----
  float rsum[8];
  #pragma unroll
  for (int g = 0; g < 8; ++g) rsum[g] = 0.f;
  for (int cb = 0; cb < NCB; ++cb) {
    const int kcol = w * WCOLS + cb * 16 + mrow;
    #pragma unroll
    for (int g = 0; g < 8; ++g) {
      int row = g + rhi;
      float p = __expf(sS[row][kcol] - sRowMax[row]);
      sS[row][kcol] = p;
      rsum[g] += p;
    }
  }
  #pragma unroll
  for (int g = 0; g < 8; ++g) {
    float s = rsum[g];
    #pragma unroll
    for (int off = 8; off >= 1; off >>= 1)
      s += __shfl_xor(s, off, 32);
    rsum[g] = s;
  }
  if (mrow == 0) {
    #pragma unroll
    for (int g = 0; g < 8; ++g) sRed[w][g + rhi] = rsum[g];
  }
  __syncthreads();
  if (t < BQ) {
    float s = 0.f;
    #pragma unroll
    for (int ww = 0; ww < NWAVE; ++ww) s += sRed[ww][t];
    sRowSum[t] = s;
  }
  __syncthreads();

  // ---- write attn_p (normalized), coalesced float4 stores ----
  {
    float* op = outP + ((size_t)bh * SEQ + q0) * SEQ;
    #pragma unroll 4
    for (int i = 0; i < (BQ * SEQ / 4) / 256; ++i) {
      int i4 = t + i * 256;
      int e = i4 * 4;
      int row = e >> 11;
      int col = e & (SEQ - 1);
      float inv = 1.0f / sRowSum[row];
      float4 pv;
      pv.x = sS[row][col + 0] * inv;
      pv.y = sS[row][col + 1] * inv;
      pv.z = sS[row][col + 2] * inv;
      pv.w = sS[row][col + 3] * inv;
      *(float4*)(op + (size_t)row * SEQ + col) = pv;
    }
  }

  // ---- O = P V, outer-K / inner-N: A-fragment conversion amortized 8x ----
  // B layout: lane = d column (nb*16+mrow) of V, K rows in table order (kbo split).
  {
    const int kbase0 = w * WCOLS;
    v8f accs[HD / 16] = {};
    for (int kc = 0; kc < WCOLS / 32; ++kc) {
      const int kofs = kbase0 + kc * 32;
      // A-fragment of P from LDS: 4x ds_load_b128 + 16 cvt
      float4 a0 = *(const float4*)&sS[mrow][kofs + kbo];
      float4 a1 = *(const float4*)&sS[mrow][kofs + kbo + 4];
      float4 a2 = *(const float4*)&sS[mrow][kofs + 16 + kbo];
      float4 a3 = *(const float4*)&sS[mrow][kofs + 16 + kbo + 4];
      v16bf ap;
      ap[0]  = (__bf16)a0.x; ap[1]  = (__bf16)a0.y; ap[2]  = (__bf16)a0.z; ap[3]  = (__bf16)a0.w;
      ap[4]  = (__bf16)a1.x; ap[5]  = (__bf16)a1.y; ap[6]  = (__bf16)a1.z; ap[7]  = (__bf16)a1.w;
      ap[8]  = (__bf16)a2.x; ap[9]  = (__bf16)a2.y; ap[10] = (__bf16)a2.z; ap[11] = (__bf16)a2.w;
      ap[12] = (__bf16)a3.x; ap[13] = (__bf16)a3.y; ap[14] = (__bf16)a3.z; ap[15] = (__bf16)a3.w;
      #pragma unroll
      for (int nb = 0; nb < HD / 16; ++nb) {
        v16bf bv;
        if constexpr (MODE >= 1) {
          const __bf16* vrow = vbs + headOff + (size_t)(kofs + kbo) * HD + nb * 16 + mrow;
          #pragma unroll
          for (int j = 0; j < 16; ++j)
            bv[j] = vrow[(size_t)(j < 8 ? j : j + 8) * HD];
        } else {
          const float* vrow = vf + headOff + (size_t)(kofs + kbo) * HD + nb * 16 + mrow;
          #pragma unroll
          for (int j = 0; j < 16; ++j)
            bv[j] = (__bf16)vrow[(size_t)(j < 8 ? j : j + 8) * HD];
        }
        accs[nb] = __builtin_amdgcn_wmma_f32_16x16x32_bf16(false, ap, false, bv,
                                                           (short)0, accs[nb], false, false);
      }
    }
    #pragma unroll
    for (int nb = 0; nb < HD / 16; ++nb) {
      #pragma unroll
      for (int g = 0; g < 8; ++g)
        atomicAdd(&sOut[g + rhi][nb * 16 + mrow], accs[nb][g]);  // ds_add_f32
    }
  }
  __syncthreads();

  // ---- normalize and store attention output ----
  {
    float* oa = outAttn + headOff + (size_t)q0 * HD;
    #pragma unroll
    for (int i = 0; i < (BQ * HD) / 256; ++i) {
      int idx = t + i * 256;
      int row = idx >> 7;
      int col = idx & (HD - 1);
      oa[(size_t)row * HD + col] = sOut[row][col] * (1.0f / sRowSum[row]);
    }
  }
}

extern "C" void kernel_launch(void* const* d_in, const int* in_sizes, int n_in,
                              void* d_out, int out_size, void* d_ws, size_t ws_size,
                              hipStream_t stream) {
  (void)in_sizes; (void)n_in; (void)out_size;
  const float* q     = (const float*)d_in[0];
  const float* k     = (const float*)d_in[1];
  const float* v     = (const float*)d_in[2];
  const int*   mask  = (const int*)d_in[3];
  const float* probs = (const float*)d_in[4];
  const float* alpha = (const float*)d_in[5];
  float* outAttn = (float*)d_out;
  float* outP    = outAttn + (size_t)BATCH * NH * SEQ * HD;

  const size_t N  = (size_t)BATCH * NH * SEQ * HD;    // per q/k/v tensor
  const size_t N2 = (size_t)BATCH * SEQ * SEQ;        // per bias tensor
  const size_t needMid  = 3 * N * sizeof(unsigned short);
  const size_t needFull = needMid + N2 * sizeof(float);

  dim3 grid(BATCH * NH * (SEQ / BQ));   // 4096 workgroups
  dim3 block(NWAVE * 32);               // 256 threads = 8 wave32

  if (ws_size >= needMid) {
    __bf16* qb = (__bf16*)d_ws;
    __bf16* kT = qb + N;                // (bh, d, l) bf16
    __bf16* vb = kT + N;
    int n4 = (int)(N / 4);
    dim3 cg((unsigned)((n4 + 255) / 256)), cb(256);
    cvt_f32_to_bf16<<<cg, cb, 0, stream>>>((const float4*)q, (v4bf*)qb, n4);
    cvt_f32_to_bf16<<<cg, cb, 0, stream>>>((const float4*)v, (v4bf*)vb, n4);
    cvt_transpose_bf16<<<dim3(SEQ / 32, HD / 32, BATCH * NH), 256, 0, stream>>>(k, kT);

    if (ws_size >= needFull) {
      float* bias = (float*)(vb + N);
      int n4b = (int)(N2 / 4);
      dim3 bg((unsigned)((n4b + 255) / 256));
      build_bias<<<bg, cb, 0, stream>>>((const float4*)probs, (const int4*)mask,
                                        alpha, (float4*)bias, n4b);
      attn_fused<2><<<grid, block, 0, stream>>>(q, k, v, qb, kT, vb, bias,
                                                mask, probs, alpha, outAttn, outP);
    } else {
      attn_fused<1><<<grid, block, 0, stream>>>(q, k, v, qb, kT, vb, nullptr,
                                                mask, probs, alpha, outAttn, outP);
    }
  } else {
    attn_fused<0><<<grid, block, 0, stream>>>(q, k, v, nullptr, nullptr, nullptr, nullptr,
                                              mask, probs, alpha, outAttn, outP);
  }
}
